// HardAttention_69638599737601
// MI455X (gfx1250) — matrix operand order repaired and checked by reading, not compiled, not run on hardware
//
#include <hip/hip_runtime.h>
#include <cstddef>
#include <cstdint>

typedef __attribute__((ext_vector_type(16))) __bf16 v16bf;
typedef __attribute__((ext_vector_type(8)))  float  v8f;
typedef __bf16 bf16x2 __attribute__((ext_vector_type(2)));

#define BB 32
#define TT 1024
#define SS 1024
#define DD 512
#define KK 1024   // 2*DD

// pack two fp32 into packed bf16 (RNE): native packed cvt builtin if present,
// else scalar __bf16 casts (clang lowers fptrunc to the hw convert on gfx1250).
__device__ __forceinline__ unsigned pack_bf16(float a, float b){
#if __has_builtin(__builtin_amdgcn_cvt_pk_bf16_f32)
  union { bf16x2 v; unsigned u; } c;
  c.v = __builtin_amdgcn_cvt_pk_bf16_f32(a, b);
  return c.u;
#else
  bf16x2 p;
  p.x = (__bf16)a;
  p.y = (__bf16)b;
  union { bf16x2 v; unsigned u; } c;
  c.v = p;
  return c.u;
#endif
}

union FragBF { v16bf v; uint4 q[2]; };

// ---------------------------------------------------------------------------
// 1) hard-align scan: one thread per batch (sequential 'inserted' counter)
// ---------------------------------------------------------------------------
__global__ void align_kernel(const int* __restrict__ input_var, int* __restrict__ jrow){
  int b = threadIdx.x;
  if (b >= BB) return;
  const int* row = input_var + b * TT;
  int* jr = jrow + b * TT;
  jr[0] = 0;                       // attn[b,0,0] = 1 always
  int inserted = 0;
  for (int i = 1; i < TT; ++i){
    int tok  = row[i];
    int flag = (tok == 1) | (tok == 2);
    int cond = (SS >= i - inserted) ? 1 : 0;
    inserted += (cond & flag);
    int j = i - inserted - 1;
    if (j < 0) j += SS;            // torch negative-index wrap
    jr[i] = cond ? j : -1;         // -1 encodes invalid row
  }
}

// ---------------------------------------------------------------------------
// 2) split W (fp32 [D][2D]) into bf16 hi/lo once; reused by all GEMM tiles
// ---------------------------------------------------------------------------
__global__ void wsplit_kernel(const float* __restrict__ W,
                              unsigned* __restrict__ Wh,
                              unsigned* __restrict__ Wl){
  int i = blockIdx.x * 256 + threadIdx.x;     // pair index; grid = D*2D/2/256
  float2 v = ((const float2*)W)[i];
  unsigned hp = pack_bf16(v.x, v.y);
  unsigned lp = pack_bf16(v.x - __uint_as_float(hp << 16),
                          v.y - __uint_as_float(hp & 0xFFFF0000u));
  Wh[i] = hp;
  Wl[i] = lp;
}

// ---------------------------------------------------------------------------
// 3) attn one-hot [B,T,S]: one block per row, float4 zero-fill + scatter 1.0
// ---------------------------------------------------------------------------
__global__ void attn_kernel(const int* __restrict__ jrow, float* __restrict__ attn){
  int row = blockIdx.x;                 // b*T + t
  int j   = jrow[row];                  // -1 => all-zero row
  int q   = threadIdx.x;                // 256 lanes cover S/4 float4s
  float4 v = make_float4(0.f, 0.f, 0.f, 0.f);
  if ((j >> 2) == q) ((float*)&v)[j & 3] = 1.0f;   // j<0: (j>>2)==-1 never hits
  ((float4*)(attn + (size_t)row * SS))[q] = v;
}

// ---------------------------------------------------------------------------
// 4) fused gather + bf16x3 WMMA GEMM + bias + tanh
//    tile: 128(M) x 64(N) per workgroup; 8 waves of 32x32 each
//    double-buffered LDS, one barrier per K-step, global prefetch overlaps WMMA
// ---------------------------------------------------------------------------
__global__ __launch_bounds__(256) void
gemm_kernel(const int* __restrict__ jrow,
            const float* __restrict__ out_in,    // output tensor [B,T,D]
            const float* __restrict__ context,   // [B,S,D]
            const unsigned short* __restrict__ Wh,
            const unsigned short* __restrict__ Wl,
            const float* __restrict__ bias,
            float* __restrict__ out)
{
  // row stride 40 ushorts (80B: 16B-aligned rows, conflict-free b128 frag loads)
  __shared__ unsigned short Ah[2][128 * 40];
  __shared__ unsigned short Al[2][128 * 40];
  __shared__ unsigned short Bh[2][64 * 40];
  __shared__ unsigned short Bl[2][64 * 40];

  const int tid    = threadIdx.x;
  const int tile_n = blockIdx.x & 7;     // 512/64  = 8 n-tiles
  const int tile_m = blockIdx.x >> 3;    // 32768/128 = 256 m-tiles
  const int m0 = tile_m * 128;
  const int n0 = tile_n * 64;

  // A staging assignment: fixed float4 slot kq, 4 rows per thread (branch-free)
  const int kq = tid & 7;
  int rowA[4];
  float ascale[4];
  const float* ctxbase[4];
  const float* outbase[4];
#pragma unroll
  for (int i = 0; i < 4; ++i){
    int r = (tid >> 3) + i * 32;
    rowA[i] = r;
    int m = m0 + r;
    int b = m >> 10;                     // m / T
    int j = jrow[m];
    int jc = (j >= 0) ? j : 0;           // clamp: always a legal address
    ascale[i] = (j >= 0) ? 1.0f : 0.0f;  // zero out invalid rows after load
    ctxbase[i] = context + ((size_t)b * SS + jc) * DD;
    outbase[i] = out_in + (size_t)m * DD;
  }
  int rowB[2];
#pragma unroll
  for (int i = 0; i < 2; ++i) rowB[i] = (tid >> 3) + i * 32;

  const int lane   = tid & 31;
  const int wv     = tid >> 5;
  const int wave_m = wv >> 1;            // 0..3
  const int wave_n = wv & 1;             // 0..1
  const int lrow   = lane & 15;
  const int hiHalf = (lane >= 16) ? 1 : 0;

  v8f acc[2][2];
#pragma unroll
  for (int mi = 0; mi < 2; ++mi)
#pragma unroll
    for (int ni = 0; ni < 2; ++ni)
      acc[mi][ni] = (v8f){0.f,0.f,0.f,0.f,0.f,0.f,0.f,0.f};

  // prefetch registers
  float4 aReg[4];
  uint2  bhReg[2], blReg[2];

  // steps 0..15: A = gathered context (mix); steps 16..31: A = output
  auto loadStep = [&](int s){
    const int kk = ((s & 15) * 32) + kq * 4;
    if (s < 16){
#pragma unroll
      for (int i = 0; i < 4; ++i){
        float4 v = *(const float4*)(ctxbase[i] + kk);
        v.x *= ascale[i]; v.y *= ascale[i]; v.z *= ascale[i]; v.w *= ascale[i];
        aReg[i] = v;
      }
    } else {
#pragma unroll
      for (int i = 0; i < 4; ++i) aReg[i] = *(const float4*)(outbase[i] + kk);
    }
    const int kw = s * 32 + kq * 4;
#pragma unroll
    for (int i = 0; i < 2; ++i){
      const size_t g = (size_t)(n0 + rowB[i]) * KK + kw;
      bhReg[i] = *(const uint2*)(Wh + g);
      blReg[i] = *(const uint2*)(Wl + g);
    }
  };

  auto storeStep = [&](int buf){
#pragma unroll
    for (int i = 0; i < 4; ++i){
      float4 v = aReg[i];
      unsigned hp0 = pack_bf16(v.x, v.y);
      unsigned hp1 = pack_bf16(v.z, v.w);
      unsigned lp0 = pack_bf16(v.x - __uint_as_float(hp0 << 16),
                               v.y - __uint_as_float(hp0 & 0xFFFF0000u));
      unsigned lp1 = pack_bf16(v.z - __uint_as_float(hp1 << 16),
                               v.w - __uint_as_float(hp1 & 0xFFFF0000u));
      int off = rowA[i] * 40 + kq * 4;
      *(uint2*)&Ah[buf][off] = make_uint2(hp0, hp1);
      *(uint2*)&Al[buf][off] = make_uint2(lp0, lp1);
    }
#pragma unroll
    for (int i = 0; i < 2; ++i){
      int off = rowB[i] * 40 + kq * 4;
      *(uint2*)&Bh[buf][off] = bhReg[i];
      *(uint2*)&Bl[buf][off] = blReg[i];
    }
  };

  loadStep(0);
  for (int s = 0; s < 32; ++s){
    const int buf = s & 1;
    storeStep(buf);
    __syncthreads();                      // single barrier per step (2 buffers)
    if (s < 31) loadStep(s + 1);          // global prefetch overlaps WMMA below

    // fragments per documented 16-bit A (16x32) / B (32x16) layouts
    FragBF fah[2], fal[2], fbh[2], fbl[2];
#pragma unroll
    for (int mi = 0; mi < 2; ++mi){
      int ar = (wave_m * 32 + mi * 16 + lrow) * 40 + hiHalf * 8;   // K 0..7 / 8..15
      fah[mi].q[0] = *(const uint4*)&Ah[buf][ar];
      fah[mi].q[1] = *(const uint4*)&Ah[buf][ar + 16];             // K 16..23 / 24..31
      fal[mi].q[0] = *(const uint4*)&Al[buf][ar];
      fal[mi].q[1] = *(const uint4*)&Al[buf][ar + 16];
    }
#pragma unroll
    for (int ni = 0; ni < 2; ++ni){
      int br = (wave_n * 32 + ni * 16 + lrow) * 40 + hiHalf * 8;   // K 0..7 / 8..15, then 16..23 / 24..31
      fbh[ni].q[0] = *(const uint4*)&Bh[buf][br];
      fbh[ni].q[1] = *(const uint4*)&Bh[buf][br + 16];
      fbl[ni].q[0] = *(const uint4*)&Bl[buf][br];
      fbl[ni].q[1] = *(const uint4*)&Bl[buf][br + 16];
    }

    // bf16x3: Ah*Bh + Ah*Bl + Al*Bh (drop lo*lo term)
#pragma unroll
    for (int mi = 0; mi < 2; ++mi)
#pragma unroll
      for (int ni = 0; ni < 2; ++ni){
        acc[mi][ni] = __builtin_amdgcn_wmma_f32_16x16x32_bf16(
            false, fah[mi].v, false, fbh[ni].v, (short)0, acc[mi][ni], false, false);
        acc[mi][ni] = __builtin_amdgcn_wmma_f32_16x16x32_bf16(
            false, fah[mi].v, false, fbl[ni].v, (short)0, acc[mi][ni], false, false);
        acc[mi][ni] = __builtin_amdgcn_wmma_f32_16x16x32_bf16(
            false, fal[mi].v, false, fbh[ni].v, (short)0, acc[mi][ni], false, false);
      }
  }

  // ---- epilogue: bias + tanh; C/D layout: VGPR v -> M = v + 8*hiHalf ----
  const int mW = m0 + wave_m * 32;
  const int nW = n0 + wave_n * 32;
#pragma unroll
  for (int mi = 0; mi < 2; ++mi){
#pragma unroll
    for (int ni = 0; ni < 2; ++ni){
      int col = nW + ni * 16 + lrow;
      float bv = bias[col];
      int r0 = mW + mi * 16 + hiHalf * 8;
#pragma unroll
      for (int v = 0; v < 8; ++v){
        float x = tanhf(acc[mi][ni][v] + bv);
        out[(size_t)(r0 + v) * DD + col] = x;
      }
    }
  }
}

// ---------------------------------------------------------------------------
extern "C" void kernel_launch(void* const* d_in, const int* in_sizes, int n_in,
                              void* d_out, int out_size, void* d_ws, size_t ws_size,
                              hipStream_t stream)
{
  const int*   input_var = (const int*)d_in[0];
  const float* output    = (const float*)d_in[1];   // [B,T,D]
  const float* context   = (const float*)d_in[2];   // [B,S,D]
  /* d_in[3] = di (unused by the reference body) */
  const float* W         = (const float*)d_in[4];   // [D,2D]
  const float* bias      = (const float*)d_in[5];   // [D]

  float* out  = (float*)d_out;                      // [B,T,D]
  float* attn = out + (size_t)BB * TT * DD;         // [B,T,S]

  char* ws = (char*)d_ws;
  int* jrow = (int*)ws;                                       // B*T ints (128 KB)
  unsigned short* Wh = (unsigned short*)(ws + 131072);        // D*2D bf16 (1 MB)
  unsigned short* Wl = Wh + (size_t)DD * KK;                  // D*2D bf16 (1 MB)

  align_kernel <<<1, 32, 0, stream>>>(input_var, jrow);
  wsplit_kernel<<<(DD * KK / 2) / 256, 256, 0, stream>>>(W, (unsigned*)Wh, (unsigned*)Wl);
  attn_kernel  <<<BB * TT, 256, 0, stream>>>(jrow, attn);
  gemm_kernel  <<<(BB * TT / 128) * (DD / 64), 256, 0, stream>>>(
      jrow, output, context, Wh, Wl, bias, out);
}
